// HUMAN_45578192945202
// MI455X (gfx1250) — compile-verified, matrix operand order checked by reading, not run on hardware
//
#include <hip/hip_runtime.h>
#include <stdint.h>

// MI455X / gfx1250, wave32. All matmuls on v_wmma_f32_16x16x32_f16; weight
// slices staged to LDS via the Tensor Data Mover when available.
//
// Input index map (setup_inputs() insertion order, params dict in insertion order):
//  0 pos 1 hks 2 weight | 3 feat_w 4 feat_b
//  5..16 c1 (e1_w,e1_b,e2_w,e2_b,c1_w,c1_b,c2_w,c2_b,n1_w,n1_b,n2_w,n2_b)
// 17..28 c2 | 29..40 c3 | 41 lin1_w 42 lin1_b 43 ln_g 44 ln_b 45 lin2_w 46 lin2_b
// 47 face[3,F] 48 vertex2face[.,2] 49 edge_index[2,E] 50 batch 51 ptr 52 face_len 53 v2f_len

typedef __attribute__((ext_vector_type(16))) _Float16 v16h;
typedef __attribute__((ext_vector_type(8)))  float    v8f;
typedef __attribute__((ext_vector_type(8)))  unsigned v8u;
typedef __attribute__((ext_vector_type(4)))  unsigned v4u;
typedef __attribute__((ext_vector_type(8)))  int      v8i;
typedef __attribute__((ext_vector_type(4)))  int      v4i;

#define CB 16
#define CNPG 4096
#define CFPG 8192
#define CEPG 24576
#define CN (CB*CNPG)
#define CF (CB*CFPG)
#define CE (CB*CEPG)

// Tensor Data Mover availability (device pass only; host pass parses fallback).
#if defined(__HIP_DEVICE_COMPILE__) && __has_builtin(__builtin_amdgcn_tensor_load_to_lds) && __has_builtin(__builtin_amdgcn_s_wait_tensorcnt)
#define GEMM_USE_TDM 1
#if __has_include(<hip/amd_detail/amd_gfx1250_TDM.h>)
#define GEMM_TDM_6ARG 1   // clang-23/therock header toolchain -> 6-arg builtin
#else
#define GEMM_TDM_6ARG 0   // ROCm 7.2 clang-22 -> 5-arg builtin
#endif
#else
#define GEMM_USE_TDM 0
#endif

// ---------------------------------------------------------------- utilities
__global__ void zero_f32_k(float* __restrict__ p, long n) {
  long i = (long)blockIdx.x * blockDim.x + threadIdx.x;
  if (i < n) p[i] = 0.0f;
}

// Pack f32 [K0,N0] row-major into fp16 k-pair-interleaved layout:
// as u32 array U[(k>>1)*N1 + n] = { f16(W[k][n]), f16(W[k+1][n]) }, zero padded.
__global__ void pack_weight_k(const float* __restrict__ src, _Float16* __restrict__ dst,
                              int K0, int N0, int K1, int N1) {
  int i = blockIdx.x * blockDim.x + threadIdx.x;
  if (i >= K1 * N1) return;
  int k = i / N1, n = i % N1;
  float v = (k < K0 && n < N0) ? src[k * N0 + n] : 0.0f;
  dst[((size_t)(k >> 1) * N1 + n) * 2 + (k & 1)] = (_Float16)v;
}

__global__ void pack_bias_k(const float* __restrict__ src, float* __restrict__ dst,
                            int N0, int N1) {
  int i = blockIdx.x * blockDim.x + threadIdx.x;
  if (i >= N1) return;
  dst[i] = (i < N0) ? src[i] : 0.0f;
}

// ------------------------------------------------- templated WMMA GEMM
// D[M,Np] = act(A[M,K] @ W[K,Np] + bias). K compile-time (full unroll),
// NT = n-tiles per wave (A-fragment reuse). 128 thr = 4 waves, wave w owns
// m-tile blockIdx.x*4+w; blockIdx.y owns NT consecutive 16-col n-tiles.
template<int K, int NT, int ACT>
__global__ __launch_bounds__(128) void wmma_gemm_t(
    const _Float16* __restrict__ A, const unsigned* __restrict__ Wp,
    const float* __restrict__ bias, _Float16* __restrict__ outH,
    float* __restrict__ outF, int lda, int ldc, int Np) {
  constexpr int KW = K / 2;        // u32 rows of packed W
  constexpr int SW = 16 * NT;      // slice width in u32 columns (= out cols)
  __shared__ unsigned Wlds[KW * SW];     // <= 192*32*4 = 24 KB

  const int tid  = threadIdx.x;
  const int lane = tid & 31;
  const int col0 = blockIdx.y * SW;

#if GEMM_USE_TDM
  if (tid < 32) {
    // D# per CDNA5 ISA 8.3/8.4: 2D tile (SW x KW u32) from tensor of width Np.
    unsigned long long ga = (unsigned long long)(size_t)(Wp + col0);
    v4u g0;
    g0[0] = 1u;                                        // count=1, user mode
    g0[1] = 0u;                                        // lds_addr: Wlds is at LDS offset 0
    g0[2] = (unsigned)(ga & 0xFFFFFFFFu);              // global_addr[31:0]
    g0[3] = (unsigned)((ga >> 32) & 0x01FFFFFFu) | (2u << 30);  // addr[56:32], type=2
    const unsigned td0 = (unsigned)Np, td1 = (unsigned)KW;
    const unsigned long long s0 = (unsigned long long)Np;
    const unsigned long long s1 = (unsigned long long)Np * (unsigned long long)KW;
    v8i g1;
    g1[0] = (int)(2u << 16);                                          // data_size=4B
    g1[1] = (int)((td0 & 0xFFFFu) << 16);                             // tensor_dim0 lo
    g1[2] = (int)(((td0 >> 16) & 0xFFFFu) | ((td1 & 0xFFFFu) << 16)); // dim0 hi | dim1 lo
    g1[3] = (int)(((td1 >> 16) & 0xFFFFu) | ((unsigned)SW << 16));    // dim1 hi | tile_dim0
    g1[4] = (int)(td1 & 0xFFFFu);                                     // tile_dim1 (tile_dim2=0)
    g1[5] = (int)(unsigned)(s0 & 0xFFFFFFFFu);                        // dim0_stride lo
    g1[6] = (int)(((s0 >> 32) & 0xFFFFu) | ((unsigned)(s1 & 0xFFFFu) << 16));
    g1[7] = (int)(unsigned)((s1 >> 16) & 0xFFFFFFFFu);
    v4i z4 = {0, 0, 0, 0};
#if GEMM_TDM_6ARG
    v8i z8 = {0, 0, 0, 0, 0, 0, 0, 0};
    __builtin_amdgcn_tensor_load_to_lds(g0, g1, z4, z4, z8, 0);
#else
    __builtin_amdgcn_tensor_load_to_lds(g0, g1, z4, z4, 0);
#endif
    __builtin_amdgcn_s_wait_tensorcnt(0);
  }
  __asm__ __volatile__("" ::: "memory");   // LDS written by TDM, not by stores
  __syncthreads();
#else
  for (int idx = tid; idx < KW * SW; idx += 128) {
    int r = idx / SW, c = idx % SW;
    Wlds[idx] = Wp[(size_t)r * Np + col0 + c];
  }
  __syncthreads();
#endif

  const int wave   = tid >> 5;
  const int tile_m = blockIdx.x * 4 + wave;
  const int lrow   = lane & 15;    // A row / D column within tile
  const int hsel   = lane >> 4;    // half-wave select (ISA 7.12.2)
  const _Float16* __restrict__ Arow = A + ((size_t)tile_m * 16 + lrow) * (size_t)lda;

  v8f acc[NT];
#pragma unroll
  for (int t = 0; t < NT; ++t) acc[t] = (v8f){0.f,0.f,0.f,0.f,0.f,0.f,0.f,0.f};

#pragma unroll
  for (int k0 = 0; k0 < K; k0 += 32) {
    v8u au;
#pragma unroll
    for (int i = 0; i < 8; ++i) {
      int kk = k0 + (((i < 4) ? (2 * i) : (16 + 2 * (i - 4))) + 8 * hsel); // even
      au[i] = *(const unsigned*)(Arow + kk);          // {A[m][kk], A[m][kk+1]}
    }
    v16h a = __builtin_bit_cast(v16h, au);
#pragma unroll
    for (int t = 0; t < NT; ++t) {
      v8u bu;
#pragma unroll
      for (int i = 0; i < 8; ++i) {
        int kk = k0 + (((i < 4) ? (2 * i) : (16 + 2 * (i - 4))) + 8 * hsel);
        bu[i] = Wlds[(kk >> 1) * SW + t * 16 + lrow]; // {W[kk][n], W[kk+1][n]}
      }
      v16h b = __builtin_bit_cast(v16h, bu);
      acc[t] = __builtin_amdgcn_wmma_f32_16x16x32_f16(
          false, a, false, b, (short)0, acc[t], false, false);
    }
  }

#pragma unroll
  for (int t = 0; t < NT; ++t) {
    const int ncol = col0 + t * 16 + lrow;
    const float bv = bias[ncol];
#pragma unroll
    for (int r = 0; r < 8; ++r) {
      int m = tile_m * 16 + r + 8 * hsel;             // D row per ISA layout
      float v = acc[t][r] + bv;
      if (ACT) v = v * __builtin_amdgcn_rcpf(1.0f + __expf(-v));  // fast SiLU
      size_t o = (size_t)m * ldc + ncol;
      if (outH) outH[o] = (_Float16)v;
      if (outF) outF[o] = v;
    }
  }
}

template<int K, int NT, int ACT>
static inline void glaunch(const _Float16* A, const unsigned* Wp, const float* bias,
                           _Float16* oH, float* oF, int M, int Np, int lda, int ldc,
                           hipStream_t s) {
  dim3 grid(M / 64, Np / (16 * NT));
  wmma_gemm_t<K, NT, ACT><<<grid, 128, 0, s>>>(A, Wp, bias, oH, oF, lda, ldc, Np);
}

static void gemm_dispatch(const _Float16* A, const unsigned* Wp, const float* bias,
                          _Float16* oH, float* oF, int M, int K, int Np,
                          int lda, int ldc, int act, hipStream_t s) {
  const bool nt2 = (Np % 32) == 0;
#define KCASE(KK)                                                                 \
  case KK:                                                                        \
    if (nt2) { if (act) glaunch<KK,2,1>(A,Wp,bias,oH,oF,M,Np,lda,ldc,s);          \
               else     glaunch<KK,2,0>(A,Wp,bias,oH,oF,M,Np,lda,ldc,s); }        \
    else     { if (act) glaunch<KK,1,1>(A,Wp,bias,oH,oF,M,Np,lda,ldc,s);          \
               else     glaunch<KK,1,0>(A,Wp,bias,oH,oF,M,Np,lda,ldc,s); }        \
    break;
  switch (K) {
    KCASE(32) KCASE(64) KCASE(96) KCASE(128) KCASE(160)
    KCASE(192) KCASE(256) KCASE(288) KCASE(384)
    default: break;
  }
#undef KCASE
}

// ------------------------------------------------------------ graph prep
__global__ void graph_stats_k(const float* __restrict__ pos,
                              float* __restrict__ centroid, float* __restrict__ radius) {
  __shared__ float sx[256], sy[256], sz[256];
  const int g = blockIdx.x, tid = threadIdx.x;
  const float* p = pos + (size_t)g * CNPG * 3;
  float ax = 0.f, ay = 0.f, az = 0.f;
  for (int i = tid; i < CNPG; i += 256) { ax += p[i*3]; ay += p[i*3+1]; az += p[i*3+2]; }
  sx[tid] = ax; sy[tid] = ay; sz[tid] = az; __syncthreads();
  for (int s = 128; s > 0; s >>= 1) {
    if (tid < s) { sx[tid]+=sx[tid+s]; sy[tid]+=sy[tid+s]; sz[tid]+=sz[tid+s]; }
    __syncthreads();
  }
  const float cx = sx[0]/CNPG, cy = sy[0]/CNPG, cz = sz[0]/CNPG;
  __syncthreads();
  float mr = 0.f;
  for (int i = tid; i < CNPG; i += 256) {
    float dx = p[i*3]-cx, dy = p[i*3+1]-cy, dz = p[i*3+2]-cz;
    mr = fmaxf(mr, sqrtf(dx*dx + dy*dy + dz*dz));
  }
  sx[tid] = mr; __syncthreads();
  for (int s = 128; s > 0; s >>= 1) {
    if (tid < s) sx[tid] = fmaxf(sx[tid], sx[tid+s]);
    __syncthreads();
  }
  if (tid == 0) { centroid[g*3]=cx; centroid[g*3+1]=cy; centroid[g*3+2]=cz; radius[g]=sx[0]; }
}

__global__ void normalize_pos_k(const float* __restrict__ pos, const float* __restrict__ centroid,
                                const float* __restrict__ radius, float* __restrict__ coord) {
  int i = blockIdx.x * blockDim.x + threadIdx.x;
  if (i >= CN) return;
  int g = i / CNPG;
  float inv = __builtin_amdgcn_rcpf(radius[g]);
  coord[i*3+0] = (pos[i*3+0] - centroid[g*3+0]) * inv;
  coord[i*3+1] = (pos[i*3+1] - centroid[g*3+1]) * inv;
  coord[i*3+2] = (pos[i*3+2] - centroid[g*3+2]) * inv;
}

__global__ void face_area_k(const float* __restrict__ coord, const int* __restrict__ face,
                            float* __restrict__ area) {
  int f = blockIdx.x * blockDim.x + threadIdx.x;
  if (f >= CF) return;
  int a = face[f], b = face[CF + f], c = face[2*CF + f];
  float ax = coord[b*3]-coord[a*3], ay = coord[b*3+1]-coord[a*3+1], az = coord[b*3+2]-coord[a*3+2];
  float bx = coord[c*3]-coord[a*3], by = coord[c*3+1]-coord[a*3+1], bz = coord[c*3+2]-coord[a*3+2];
  float cx = ay*bz - az*by, cy = az*bx - ax*bz, cz = ax*by - ay*bx;
  area[f] = 0.5f * sqrtf(cx*cx + cy*cy + cz*cz);
}

__global__ void varea_scatter_k(const int* __restrict__ v2f, const float* __restrict__ area,
                                float* __restrict__ vsum, float* __restrict__ vcnt) {
  int i = blockIdx.x * blockDim.x + threadIdx.x;
  if (i >= CB * 3 * CFPG) return;
  int g  = i / (3 * CFPG);
  int vg = v2f[(size_t)i*2]   + g * CNPG;
  int fg = v2f[(size_t)i*2+1] + g * CFPG;
  atomicAdd(&vsum[vg], area[fg]);
  atomicAdd(&vcnt[vg], 1.0f);
}

__global__ void feat_gather_k(const float* __restrict__ vsum, const float* __restrict__ vcnt,
                              const float* __restrict__ hks, _Float16* __restrict__ featIn) {
  int i = blockIdx.x * blockDim.x + threadIdx.x;
  if (i >= CN) return;
  float c = fmaxf(vcnt[i], 1.0f);
  _Float16* d = featIn + (size_t)i * 32;
  d[0] = (_Float16)(vsum[i] / c);
#pragma unroll
  for (int j = 0; j < 9; ++j) d[1 + j] = (_Float16)hks[(size_t)i*9 + j];
#pragma unroll
  for (int j = 10; j < 32; ++j) d[j] = (_Float16)0.0f;
}

// ------------------------------------------------------------ EGCL glue
__global__ void edge_gather_k(const _Float16* __restrict__ h, int din,
                              const float* __restrict__ coord,
                              const int* __restrict__ row, const int* __restrict__ col,
                              const float* __restrict__ w, _Float16* __restrict__ Ain, int K1) {
  int e = blockIdx.x * blockDim.x + threadIdx.x;
  if (e >= CE) return;
  int r = row[e], c = col[e];
  float dx = coord[r*3]-coord[c*3], dy = coord[r*3+1]-coord[c*3+1], dz = coord[r*3+2]-coord[c*3+2];
  float rad = dx*dx + dy*dy + dz*dz;
  unsigned* d32 = (unsigned*)(Ain + (size_t)e * K1);
  const unsigned* hr = (const unsigned*)(h + (size_t)r * din);
  const unsigned* hc = (const unsigned*)(h + (size_t)c * din);
  const int dw = din >> 1;
  for (int j = 0; j < dw; ++j) d32[j] = hr[j];
  for (int j = 0; j < dw; ++j) d32[dw + j] = hc[j];
  union { _Float16 h2[2]; unsigned u; } t;
  t.h2[0] = (_Float16)rad; t.h2[1] = (_Float16)w[e];
  d32[din] = t.u;
  for (int j = din + 1; j < (K1 >> 1); ++j) d32[j] = 0u;
}

__global__ void coord_scatter_k(const float* __restrict__ coord, const int* __restrict__ row,
                                const int* __restrict__ col, const float* __restrict__ cval,
                                float* __restrict__ acc4) {
  int e = blockIdx.x * blockDim.x + threadIdx.x;
  if (e >= CE) return;
  int r = row[e], c = col[e];
  float cc = cval[(size_t)e * 16];
  atomicAdd(&acc4[r*4+0], (coord[r*3+0]-coord[c*3+0]) * cc);
  atomicAdd(&acc4[r*4+1], (coord[r*3+1]-coord[c*3+1]) * cc);
  atomicAdd(&acc4[r*4+2], (coord[r*3+2]-coord[c*3+2]) * cc);
  atomicAdd(&acc4[r*4+3], 1.0f);
}

__global__ void coord_apply_k(float* __restrict__ coord, const float* __restrict__ acc4) {
  int i = blockIdx.x * blockDim.x + threadIdx.x;
  if (i >= CN) return;
  float inv = __builtin_amdgcn_rcpf(fmaxf(acc4[i*4+3], 1.0f));
  coord[i*3+0] += acc4[i*4+0] * inv;
  coord[i*3+1] += acc4[i*4+1] * inv;
  coord[i*3+2] += acc4[i*4+2] * inv;
}

__global__ void agg_scatter_k(const _Float16* __restrict__ m, int dh,
                              const int* __restrict__ row, float* __restrict__ agg, long total) {
  long i = (long)blockIdx.x * blockDim.x + threadIdx.x;
  if (i >= total) return;
  int e = (int)(i / dh), j = (int)(i % dh);
  atomicAdd(&agg[(size_t)row[e] * dh + j], (float)m[(size_t)e * dh + j]);
}

__global__ void node_gather_k(const _Float16* __restrict__ h, int din,
                              const float* __restrict__ agg, int dh,
                              _Float16* __restrict__ out, long total) {
  long i = (long)blockIdx.x * blockDim.x + threadIdx.x;
  if (i >= total) return;
  int K = din + dh;
  long n = i / K; int j = (int)(i % K);
  out[i] = (j < din) ? h[n * din + j] : (_Float16)agg[n * dh + (j - din)];
}

// ------------------------------------------------------------ readout
__global__ void face_scatter_k(const int* __restrict__ v2f, const _Float16* __restrict__ x,
                               float* __restrict__ xfsum, long total) {
  long i = (long)blockIdx.x * blockDim.x + threadIdx.x;
  if (i >= total) return;
  long ent = i >> 8; int j = (int)(i & 255);
  int g  = (int)(ent / (3 * CFPG));
  int vg = v2f[ent*2]   + g * CNPG;
  int fg = v2f[ent*2+1] + g * CFPG;
  atomicAdd(&xfsum[(size_t)fg * 256 + j], (float)x[(size_t)vg * 256 + j]);
}

__global__ void face_mean_cvt_k(const float* __restrict__ xfsum, _Float16* __restrict__ faceIn, long n) {
  long i = (long)blockIdx.x * blockDim.x + threadIdx.x;
  if (i >= n) return;
  faceIn[i] = (_Float16)(xfsum[i] * (1.0f / 3.0f));   // each face has exactly 3 v2f entries
}

__global__ void ln_stats_k(const float* __restrict__ y, float* __restrict__ stats) {
  __shared__ float s1[256], s2[256];
  const int g = blockIdx.x, tid = threadIdx.x;
  const float* p = y + (size_t)g * CFPG * 256;
  float a = 0.f, b = 0.f;
  for (long i = tid; i < (long)CFPG * 256; i += 256) { float v = p[i]; a += v; b += v * v; }
  s1[tid] = a; s2[tid] = b; __syncthreads();
  for (int s = 128; s > 0; s >>= 1) {
    if (tid < s) { s1[tid] += s1[tid+s]; s2[tid] += s2[tid+s]; }
    __syncthreads();
  }
  if (tid == 0) { stats[g*2] = s1[0]; stats[g*2+1] = s2[0]; }
}

__global__ void ln_apply_k(const float* __restrict__ y, const float* __restrict__ stats,
                           const float* __restrict__ gam, const float* __restrict__ bet,
                           _Float16* __restrict__ out, long total) {
  long i = (long)blockIdx.x * blockDim.x + threadIdx.x;
  if (i >= total) return;
  long f = i >> 8; int j = (int)(i & 255);
  int g = (int)(f / CFPG);
  const float cnt = (float)CFPG * 256.0f;
  float mu  = stats[g*2] / cnt;
  float var = stats[g*2+1] / cnt - mu * mu;
  float v = (y[i] - mu) * rsqrtf(var + 1e-5f) * gam[j] + bet[j];
  out[i] = (_Float16)fmaxf(v, 0.0f);
}

__global__ void logsoftmax_k(const float* __restrict__ z, float* __restrict__ out) {
  int f = blockIdx.x * blockDim.x + threadIdx.x;
  if (f >= CF) return;
  float v[8], mx = -1e30f;
#pragma unroll
  for (int j = 0; j < 8; ++j) { v[j] = z[(size_t)f * 16 + j]; mx = fmaxf(mx, v[j]); }
  float s = 0.f;
#pragma unroll
  for (int j = 0; j < 8; ++j) s += __expf(v[j] - mx);
  float l = __logf(s);
#pragma unroll
  for (int j = 0; j < 8; ++j) out[(size_t)f * 8 + j] = v[j] - mx - l;
}

// ---------------------------------------------------------------- host
extern "C" void kernel_launch(void* const* d_in, const int* in_sizes, int n_in,
                              void* d_out, int out_size, void* d_ws, size_t ws_size,
                              hipStream_t stream) {
  (void)in_sizes; (void)n_in; (void)out_size; (void)ws_size;
  const float* pos  = (const float*)d_in[0];
  const float* hks  = (const float*)d_in[1];
  const float* wgt  = (const float*)d_in[2];
  const int*   face = (const int*)d_in[47];
  const int*   v2f  = (const int*)d_in[48];
  const int*   eidx = (const int*)d_in[49];
  const int*   erow = eidx;
  const int*   ecol = eidx + CE;
  const float* ln_g = (const float*)d_in[43];
  const float* ln_b = (const float*)d_in[44];

  char* ws = (char*)d_ws;
  size_t off = 0;
  auto alloc = [&](size_t bytes) -> void* {
    off = (off + 255) & ~(size_t)255;
    void* p = ws + off;
    off += bytes;
    return p;
  };

  float*    coord   = (float*)alloc((size_t)CN * 3 * 4);
  float*    centroid= (float*)alloc(CB * 3 * 4);
  float*    radius  = (float*)alloc(CB * 4);
  float*    areaF   = (float*)alloc((size_t)CF * 4);
  float*    vsum    = (float*)alloc((size_t)CN * 4);
  float*    vcnt    = (float*)alloc((size_t)CN * 4);
  float*    acc4    = (float*)alloc((size_t)CN * 4 * 4);
  float*    aggF    = (float*)alloc((size_t)CN * 128 * 4);
  float*    cval    = (float*)alloc((size_t)CE * 16 * 4);
  _Float16* hA      = (_Float16*)alloc((size_t)CN * 256 * 2);
  _Float16* hB      = (_Float16*)alloc((size_t)CN * 256 * 2);
  _Float16* ebufA   = (_Float16*)alloc((size_t)CE * 288 * 2);
  _Float16* ebufB   = (_Float16*)alloc((size_t)CE * 288 * 2);
  _Float16* nodeIn  = (_Float16*)alloc((size_t)CN * 384 * 2);
  _Float16* nodeH   = (_Float16*)alloc((size_t)CN * 128 * 2);
  float*    xfsum   = (float*)alloc((size_t)CF * 256 * 4);
  _Float16* faceIn  = (_Float16*)alloc((size_t)CF * 256 * 2);
  float*    lnst    = (float*)alloc(CB * 2 * 4);
  float*    zF      = (float*)alloc((size_t)CF * 16 * 4);
  float*    yF      = xfsum;    // free after face_mean_cvt -> alias
  _Float16* ylnH    = faceIn;   // free after lin1 -> alias
  _Float16* featIn  = nodeIn;   // free early -> alias

  struct PW { unsigned* w; float* b; };
  auto packW = [&](int wi, int K0, int N0, int K1, int N1) -> PW {
    PW r;
    r.w = (unsigned*)alloc((size_t)K1 * N1 * 2);
    r.b = (float*)alloc((size_t)N1 * 4);
    pack_weight_k<<<(K1 * N1 + 255) / 256, 256, 0, stream>>>(
        (const float*)d_in[wi], (_Float16*)r.w, K0, N0, K1, N1);
    pack_bias_k<<<(N1 + 63) / 64, 64, 0, stream>>>((const float*)d_in[wi + 1], r.b, N0, N1);
    return r;
  };

  const int dinL[3] = {32, 64, 128}, dhL[3] = {32, 64, 128}, doutL[3] = {64, 128, 256};
  const int baseL[3] = {5, 17, 29};
  PW feat = packW(3, 10, 32, 32, 32);
  PW e1[3], e2[3], c1[3], c2[3], n1[3], n2[3];
  for (int l = 0; l < 3; ++l) {
    int din = dinL[l], dh = dhL[l], dout = doutL[l], base = baseL[l];
    int K1 = ((2 * din + 2) + 31) / 32 * 32;
    e1[l] = packW(base + 0,  2 * din + 2, dh,   K1,       dh);
    e2[l] = packW(base + 2,  dh,          dh,   dh,       dh);
    c1[l] = packW(base + 4,  dh,          dh,   dh,       dh);
    c2[l] = packW(base + 6,  dh,          1,    dh,       16);
    n1[l] = packW(base + 8,  din + dh,    dh,   din + dh, dh);
    n2[l] = packW(base + 10, dh,          dout, dh,       dout);
  }
  PW lin1 = packW(41, 256, 256, 256, 256);
  PW lin2 = packW(45, 256, 8,   256, 16);

  auto gemm = [&](const _Float16* A, PW w, _Float16* oH, float* oF,
                  int M, int K, int Np, int lda, int ldc, int act) {
    gemm_dispatch(A, w.w, w.b, oH, oF, M, K, Np, lda, ldc, act, stream);
  };
  auto zero = [&](float* p, long n) {
    zero_f32_k<<<(int)((n + 255) / 256), 256, 0, stream>>>(p, n);
  };

  // ---- pos_normalize, face areas, vertex areas
  graph_stats_k<<<CB, 256, 0, stream>>>(pos, centroid, radius);
  normalize_pos_k<<<CN / 256, 256, 0, stream>>>(pos, centroid, radius, coord);
  face_area_k<<<CF / 256, 256, 0, stream>>>(coord, face, areaF);
  zero(vsum, CN); zero(vcnt, CN);
  varea_scatter_k<<<(CB * 3 * CFPG) / 256, 256, 0, stream>>>(v2f, areaF, vsum, vcnt);

  // ---- feature linear (10 -> 32)
  feat_gather_k<<<CN / 256, 256, 0, stream>>>(vsum, vcnt, hks, featIn);
  gemm(featIn, feat, hA, nullptr, CN, 32, 32, 32, 32, 0);

  // ---- 3 EGCL layers
  _Float16* hcur = hA;
  _Float16* hnxt = hB;
  for (int l = 0; l < 3; ++l) {
    int din = dinL[l], dh = dhL[l], dout = doutL[l];
    int K1 = ((2 * din + 2) + 31) / 32 * 32;
    edge_gather_k<<<CE / 256, 256, 0, stream>>>(hcur, din, coord, erow, ecol, wgt, ebufA, K1);
    gemm(ebufA, e1[l], ebufB, nullptr, CE, K1, dh, K1, dh, 1);   // silu
    gemm(ebufB, e2[l], ebufA, nullptr, CE, dh, dh, dh, dh, 1);   // m (silu) in ebufA
    gemm(ebufA, c1[l], ebufB, nullptr, CE, dh, dh, dh, dh, 1);   // silu
    gemm(ebufB, c2[l], nullptr, cval,  CE, dh, 16, dh, 16, 0);   // c scalar (col 0)
    zero(acc4, (long)CN * 4);
    coord_scatter_k<<<CE / 256, 256, 0, stream>>>(coord, erow, ecol, cval, acc4);
    coord_apply_k<<<CN / 256, 256, 0, stream>>>(coord, acc4);
    zero(aggF, (long)CN * dh);
    agg_scatter_k<<<(int)(((long)CE * dh) / 256), 256, 0, stream>>>(ebufA, dh, erow, aggF, (long)CE * dh);
    node_gather_k<<<(int)(((long)CN * (din + dh)) / 256), 256, 0, stream>>>(
        hcur, din, aggF, dh, nodeIn, (long)CN * (din + dh));
    gemm(nodeIn, n1[l], nodeH, nullptr, CN, din + dh, dh, din + dh, dh, 1);
    gemm(nodeH,  n2[l], hnxt,  nullptr, CN, dh, dout, dh, dout, 0);
    _Float16* t = hcur; hcur = hnxt; hnxt = t;
  }

  // ---- vertex -> face mean, lin1, graph LayerNorm, relu, lin2, log_softmax
  zero(xfsum, (long)CF * 256);
  face_scatter_k<<<(int)(((long)CB * 3 * CFPG * 256) / 256), 256, 0, stream>>>(
      v2f, hcur, xfsum, (long)CB * 3 * CFPG * 256);
  face_mean_cvt_k<<<(int)(((long)CF * 256) / 256), 256, 0, stream>>>(xfsum, faceIn, (long)CF * 256);
  gemm(faceIn, lin1, nullptr, yF, CF, 256, 256, 256, 256, 0);
  ln_stats_k<<<CB, 256, 0, stream>>>(yF, lnst);
  ln_apply_k<<<(int)(((long)CF * 256) / 256), 256, 0, stream>>>(yF, lnst, ln_g, ln_b, ylnH, (long)CF * 256);
  gemm(ylnH, lin2, nullptr, zF, CF, 256, 16, 256, 16, 0);
  logsoftmax_k<<<CF / 256, 256, 0, stream>>>(zF, (float*)d_out);
}
